// AttentiveFPSimModel_31525059953094
// MI455X (gfx1250) — matrix operand order repaired and checked by reading, not compiled, not run on hardware
//
#include <hip/hip_runtime.h>
#include <hip/hip_bf16.h>

typedef __attribute__((ext_vector_type(16))) _Float16 v16h;
typedef __attribute__((ext_vector_type(8)))  _Float16 v8h;
typedef __attribute__((ext_vector_type(8)))  float    v8f;

union F16x16 { v16h v; v8h h[2]; };

#define NN 8192
#define DD 256

__device__ __forceinline__ float wave_sum(float x) {
#pragma unroll
  for (int off = 16; off > 0; off >>= 1) x += __shfl_xor(x, off, 32);
  return x;
}

__device__ __forceinline__ float gelu_exact(float x) {
  return 0.5f * x * (1.0f + erff(x * 0.70710678118654752f));
}

// ---------------- packing kernels ----------------
__global__ void cvt_f16_kernel(const float* __restrict__ src, _Float16* __restrict__ dst, int n) {
  int i = blockIdx.x * blockDim.x + threadIdx.x;
  if (i < n) dst[i] = (_Float16)src[i];
}

// Wt[n][k] = W[k][n]  (W is [K,N] row-major)
__global__ void pack_wT_kernel(const float* __restrict__ W, _Float16* __restrict__ Wt, int K, int N) {
  int i = blockIdx.x * blockDim.x + threadIdx.x;
  if (i < K * N) {
    int n = i / K, k = i % K;
    Wt[i] = (_Float16)W[(size_t)k * N + n];
  }
}

// ---------------- generic WMMA GEMM: out = act(A@Bt^T + bias) ----------------
// A: [M,K] f16 row-major. Bt: [N,K] f16 row-major (i.e. B transposed). bias: [N] f32.
// One wave (32 threads) computes a 16x16 tile.
__global__ void gemm16_kernel(const _Float16* __restrict__ A, const _Float16* __restrict__ Bt,
                              const float* __restrict__ bias, float* __restrict__ C,
                              _Float16* __restrict__ Ch, int M, int N, int K, int act) {
  const int lane = threadIdx.x;
  const int half = lane >> 4;
  const int l15  = lane & 15;
  const int col0 = blockIdx.x * 16;
  const int row0 = blockIdx.y * 16;

  v8f c = {};
  const _Float16* arow = A  + (size_t)(row0 + l15) * K + half * 8;
  const _Float16* brow = Bt + (size_t)(col0 + l15) * K + half * 8;

  for (int k0 = 0; k0 < K; k0 += 32) {
    F16x16 af, bf;
    af.h[0] = *(const v8h*)(arow + k0);        // K = k0+kb .. +7
    af.h[1] = *(const v8h*)(arow + k0 + 16);   // K = k0+kb+16 .. +23
    bf.h[0] = *(const v8h*)(brow + k0);        // K = k0+kb .. +7
    bf.h[1] = *(const v8h*)(brow + k0 + 16);
    c = __builtin_amdgcn_wmma_f32_16x16x32_f16(false, af.v, false, bf.v, (short)0, c, false, false);
  }

  const float bc = bias[col0 + l15];
#pragma unroll
  for (int r = 0; r < 8; ++r) {
    const int row = row0 + r + half * 8;
    float v2 = c[r] + bc;
    if (act) v2 = gelu_exact(v2);
    const size_t o = (size_t)row * N + col0 + l15;
    if (C)  C[o]  = v2;
    if (Ch) Ch[o] = (_Float16)v2;
  }
}

// ---------------- fused sim = Xp@Xp^T/16 + E, streaming per-row top-8 ----------------
// One wave per 16-row block; sweeps all 8192 columns in 16-wide tiles.
__global__ void sim_topk_kernel(const _Float16* __restrict__ Xph, const float* __restrict__ E,
                                int* __restrict__ topk) {
  const int lane = threadIdx.x;
  const int half = lane >> 4;
  const int l15  = lane & 15;
  const int row0 = blockIdx.x * 16;

  __shared__ float thr[16];
  __shared__ float tv[16][8];
  __shared__ int   ti[16][8];
  __shared__ float tileS[16][16];

  if (lane < 16) {
    thr[lane] = -1e30f;
#pragma unroll
    for (int j = 0; j < 8; ++j) { tv[lane][j] = -1e30f; ti[lane][j] = 0; }
  }
  __syncthreads();

  // Pin A fragments (16 rows x 256 K) in registers for the whole sweep.
  F16x16 afr[8];
  {
    const _Float16* arow = Xph + (size_t)(row0 + l15) * DD + half * 8;
#pragma unroll
    for (int s = 0; s < 8; ++s) {
      afr[s].h[0] = *(const v8h*)(arow + s * 32);
      afr[s].h[1] = *(const v8h*)(arow + s * 32 + 16);
    }
  }

  float thl[8];
#pragma unroll
  for (int r = 0; r < 8; ++r) thl[r] = thr[r + half * 8];

  for (int j0 = 0; j0 < NN; j0 += 16) {
    v8f c = {};
    const _Float16* brow = Xph + (size_t)(j0 + l15) * DD + half * 8;
#pragma unroll
    for (int s = 0; s < 8; ++s) {
      F16x16 bf;
      bf.h[0] = *(const v8h*)(brow + s * 32);
      bf.h[1] = *(const v8h*)(brow + s * 32 + 16);
      c = __builtin_amdgcn_wmma_f32_16x16x32_f16(false, afr[s].v, false, bf.v, (short)0, c, false, false);
    }
    const int col = j0 + l15;
    float fv[8];
    bool cand = false;
#pragma unroll
    for (int r = 0; r < 8; ++r) {
      const int tr = r + half * 8;
      float f = c[r] * 0.0625f + E[(size_t)(row0 + tr) * NN + col];
      fv[r] = f;
      cand |= (f > thl[r]);
    }
    if (__ballot((int)cand) != 0ull) {   // rare slow path: insert into sorted top-8
#pragma unroll
      for (int r = 0; r < 8; ++r) tileS[r + half * 8][l15] = fv[r];
      __syncthreads();
      if (lane < 16) {
        float t8 = tv[lane][7];
        for (int cc = 0; cc < 16; ++cc) {
          float v2 = tileS[lane][cc];
          if (v2 > t8) {
            int pos = 7;
            while (pos > 0 && v2 > tv[lane][pos - 1]) --pos;
            for (int s2 = 7; s2 > pos; --s2) { tv[lane][s2] = tv[lane][s2 - 1]; ti[lane][s2] = ti[lane][s2 - 1]; }
            tv[lane][pos] = v2; ti[lane][pos] = j0 + cc;
            t8 = tv[lane][7];
          }
        }
        thr[lane] = t8;
      }
      __syncthreads();
#pragma unroll
      for (int r = 0; r < 8; ++r) thl[r] = thr[r + half * 8];
    }
  }
  __syncthreads();
  if (lane < 16) {
#pragma unroll
    for (int j = 0; j < 8; ++j) topk[(size_t)(row0 + lane) * 8 + j] = ti[lane][j];
  }
}

// ---------------- sparse masked attention + residual + LN1 ----------------
__global__ void attn_ln_kernel(const float* __restrict__ X, const float* __restrict__ Q,
                               const float* __restrict__ Km, const float* __restrict__ V,
                               const int* __restrict__ topk,
                               const float* __restrict__ g, const float* __restrict__ b,
                               float* __restrict__ h, _Float16* __restrict__ hh) {
  const int i = blockIdx.x * 8 + (threadIdx.x >> 5);
  const int lane = threadIdx.x & 31;

  int nb[9]; bool selfin = false;
#pragma unroll
  for (int j = 0; j < 8; ++j) { int t = topk[i * 8 + j]; nb[j] = t; selfin |= (t == i); }
  nb[8] = i;

  float qv[8];
  const float* qr = Q + (size_t)i * DD + lane * 8;
#pragma unroll
  for (int t = 0; t < 8; ++t) qv[t] = qr[t];

  float sc[9];
#pragma unroll
  for (int j = 0; j < 9; ++j) {
    const float* kr = Km + (size_t)nb[j] * DD + lane * 8;
    float d = 0.f;
#pragma unroll
    for (int t = 0; t < 8; ++t) d += qv[t] * kr[t];
    sc[j] = wave_sum(d) * 0.0625f;
  }
  if (selfin) sc[8] = -1e30f;   // avoid double-counting self

  float mx = sc[0];
#pragma unroll
  for (int j = 1; j < 9; ++j) mx = fmaxf(mx, sc[j]);
  float aw[9]; float se = 0.f;
#pragma unroll
  for (int j = 0; j < 9; ++j) { aw[j] = expf(sc[j] - mx); se += aw[j]; }
  const float inv = 1.f / se;

  float mv[8] = {0.f, 0.f, 0.f, 0.f, 0.f, 0.f, 0.f, 0.f};
#pragma unroll
  for (int j = 0; j < 9; ++j) {
    const float a = aw[j] * inv;
    const float* vr = V + (size_t)nb[j] * DD + lane * 8;
#pragma unroll
    for (int t = 0; t < 8; ++t) mv[t] += a * vr[t];
  }

  const float* xr = X + (size_t)i * DD + lane * 8;
  float xv[8]; float s = 0.f;
#pragma unroll
  for (int t = 0; t < 8; ++t) { xv[t] = xr[t] + mv[t]; s += xv[t]; }
  const float mean = wave_sum(s) * (1.f / 256.f);
  float vs = 0.f;
#pragma unroll
  for (int t = 0; t < 8; ++t) { float d2 = xv[t] - mean; vs += d2 * d2; }
  const float rstd = rsqrtf(wave_sum(vs) * (1.f / 256.f) + 1e-5f);
#pragma unroll
  for (int t = 0; t < 8; ++t) {
    const int d3 = lane * 8 + t;
    const float o = (xv[t] - mean) * rstd * g[d3] + b[d3];
    h[(size_t)i * DD + d3]  = o;
    hh[(size_t)i * DD + d3] = (_Float16)o;
  }
}

// ---------------- h2 = LN(A + B) ----------------
__global__ void add_ln_kernel(const float* __restrict__ A, const float* __restrict__ B,
                              const float* __restrict__ g, const float* __restrict__ b,
                              float* __restrict__ hf, _Float16* __restrict__ hh) {
  const int i = blockIdx.x * 8 + (threadIdx.x >> 5);
  const int lane = threadIdx.x & 31;
  const size_t base = (size_t)i * DD + lane * 8;
  float xv[8]; float s = 0.f;
#pragma unroll
  for (int t = 0; t < 8; ++t) { xv[t] = A[base + t] + B[base + t]; s += xv[t]; }
  const float mean = wave_sum(s) * (1.f / 256.f);
  float vs = 0.f;
#pragma unroll
  for (int t = 0; t < 8; ++t) { float d2 = xv[t] - mean; vs += d2 * d2; }
  const float rstd = rsqrtf(wave_sum(vs) * (1.f / 256.f) + 1e-5f);
#pragma unroll
  for (int t = 0; t < 8; ++t) {
    const int d3 = lane * 8 + t;
    const float o = (xv[t] - mean) * rstd * g[d3] + b[d3];
    if (hf) hf[base + t] = o;
    hh[base + t] = (_Float16)o;
  }
}

// ---------------- final head: out[i] = y2[i] . Wh3 + bh3 ----------------
__global__ void head_kernel(const float* __restrict__ Y, const float* __restrict__ w,
                            const float* __restrict__ b3, float* __restrict__ out) {
  const int i = blockIdx.x * 8 + (threadIdx.x >> 5);
  const int lane = threadIdx.x & 31;
  const float* yr = Y + (size_t)i * DD + lane * 8;
  float s = 0.f;
#pragma unroll
  for (int t = 0; t < 8; ++t) s += yr[t] * w[lane * 8 + t];
  s = wave_sum(s);
  if (lane == 0) out[i] = s + b3[0];
}

// ---------------- launch ----------------
extern "C" void kernel_launch(void* const* d_in, const int* in_sizes, int n_in,
                              void* d_out, int out_size, void* d_ws, size_t ws_size,
                              hipStream_t stream) {
  (void)in_sizes; (void)n_in; (void)out_size; (void)ws_size;

  const float* X   = (const float*)d_in[0];
  const float* E   = (const float*)d_in[1];
  const float* Wg  = (const float*)d_in[2];  const float* bg  = (const float*)d_in[3];
  const float* Wq  = (const float*)d_in[4];  const float* bq  = (const float*)d_in[5];
  const float* Wk  = (const float*)d_in[6];  const float* bk  = (const float*)d_in[7];
  const float* Wv  = (const float*)d_in[8];  const float* bv  = (const float*)d_in[9];
  const float* g1  = (const float*)d_in[10]; const float* bn1 = (const float*)d_in[11];
  const float* W1  = (const float*)d_in[12]; const float* b1  = (const float*)d_in[13];
  const float* W2  = (const float*)d_in[14]; const float* b2  = (const float*)d_in[15];
  const float* g2  = (const float*)d_in[16]; const float* bn2 = (const float*)d_in[17];
  const float* Wh1 = (const float*)d_in[18]; const float* bh1 = (const float*)d_in[19];
  const float* Wh2 = (const float*)d_in[20]; const float* bh2 = (const float*)d_in[21];
  const float* Wh3 = (const float*)d_in[22]; const float* bh3 = (const float*)d_in[23];
  float* out = (float*)d_out;

  char* ws = (char*)d_ws;
  size_t off = 0;
  auto alloc = [&](size_t bytes) -> char* {
    char* p = ws + off;
    off += (bytes + 255) & ~(size_t)255;
    return p;
  };
  const int M = NN, D = DD;

  _Float16* Xh   = (_Float16*)alloc((size_t)M * D * 2);
  _Float16* XpH  = (_Float16*)alloc((size_t)M * D * 2);
  _Float16* WgT  = (_Float16*)alloc((size_t)256 * 256 * 2);
  _Float16* WqT  = (_Float16*)alloc((size_t)256 * 256 * 2);
  _Float16* WkT  = (_Float16*)alloc((size_t)256 * 256 * 2);
  _Float16* WvT  = (_Float16*)alloc((size_t)256 * 256 * 2);
  _Float16* W1T  = (_Float16*)alloc((size_t)256 * 256 * 2);
  _Float16* W2T  = (_Float16*)alloc((size_t)256 * 256 * 2);
  _Float16* Wh1T = (_Float16*)alloc((size_t)256 * 512 * 2);
  _Float16* Wh2T = (_Float16*)alloc((size_t)512 * 256 * 2);
  float*    qB   = (float*)alloc((size_t)M * D * 4);
  float*    kB   = (float*)alloc((size_t)M * D * 4);
  float*    vB   = (float*)alloc((size_t)M * D * 4);
  int*      topk = (int*)alloc((size_t)M * 8 * 4);
  float*    hB   = (float*)alloc((size_t)M * D * 4);
  _Float16* hH   = (_Float16*)alloc((size_t)M * D * 2);
  _Float16* z1H  = (_Float16*)alloc((size_t)M * D * 2);
  float*    zB   = (float*)alloc((size_t)M * D * 4);
  _Float16* h2H  = (_Float16*)alloc((size_t)M * D * 2);
  _Float16* y1H  = (_Float16*)alloc((size_t)M * 512 * 2);
  float*    y2B  = (float*)alloc((size_t)M * D * 4);

  const int nXD = M * D;

  // pack inputs to f16
  cvt_f16_kernel<<<(nXD + 255) / 256, 256, 0, stream>>>(X, Xh, nXD);
  pack_wT_kernel<<<(256 * 256 + 255) / 256, 256, 0, stream>>>(Wg, WgT, 256, 256);
  pack_wT_kernel<<<(256 * 256 + 255) / 256, 256, 0, stream>>>(Wq, WqT, 256, 256);
  pack_wT_kernel<<<(256 * 256 + 255) / 256, 256, 0, stream>>>(Wk, WkT, 256, 256);
  pack_wT_kernel<<<(256 * 256 + 255) / 256, 256, 0, stream>>>(Wv, WvT, 256, 256);
  pack_wT_kernel<<<(256 * 256 + 255) / 256, 256, 0, stream>>>(W1, W1T, 256, 256);
  pack_wT_kernel<<<(256 * 256 + 255) / 256, 256, 0, stream>>>(W2, W2T, 256, 256);
  pack_wT_kernel<<<(256 * 512 + 255) / 256, 256, 0, stream>>>(Wh1, Wh1T, 256, 512);
  pack_wT_kernel<<<(512 * 256 + 255) / 256, 256, 0, stream>>>(Wh2, Wh2T, 512, 256);

  // projections: Xp (f16 only), q, k, v (f32)
  dim3 g256(256 / 16, M / 16);
  gemm16_kernel<<<g256, 32, 0, stream>>>(Xh, WgT, bg, nullptr, XpH, M, 256, 256, 0);
  gemm16_kernel<<<g256, 32, 0, stream>>>(Xh, WqT, bq, qB, nullptr, M, 256, 256, 0);
  gemm16_kernel<<<g256, 32, 0, stream>>>(Xh, WkT, bk, kB, nullptr, M, 256, 256, 0);
  gemm16_kernel<<<g256, 32, 0, stream>>>(Xh, WvT, bv, vB, nullptr, M, 256, 256, 0);

  // dense similarity + streamed top-8
  sim_topk_kernel<<<M / 16, 32, 0, stream>>>(XpH, E, topk);

  // sparse attention + residual + LN1
  attn_ln_kernel<<<M / 8, 256, 0, stream>>>(X, qB, kB, vB, topk, g1, bn1, hB, hH);

  // FFN: z = gelu(h@W1+b1)@W2+b2 ; h2 = LN(h+z)
  gemm16_kernel<<<g256, 32, 0, stream>>>(hH, W1T, b1, nullptr, z1H, M, 256, 256, 1);
  gemm16_kernel<<<g256, 32, 0, stream>>>(z1H, W2T, b2, zB, nullptr, M, 256, 256, 0);
  add_ln_kernel<<<M / 8, 256, 0, stream>>>(hB, zB, g2, bn2, nullptr, h2H);

  // head
  dim3 g512(512 / 16, M / 16);
  gemm16_kernel<<<g512, 32, 0, stream>>>(h2H, Wh1T, bh1, nullptr, y1H, M, 512, 256, 1);
  gemm16_kernel<<<g256, 32, 0, stream>>>(y1H, Wh2T, bh2, y2B, nullptr, M, 256, 512, 1);
  head_kernel<<<M / 8, 256, 0, stream>>>(y2B, Wh3, bh3, out);
}
